// CIM_Linear_103079215221
// MI455X (gfx1250) — 8-bit matrix operand order repaired and checked by reading, compiled, not run on hardware
//
#include <hip/hip_runtime.h>
#include <hip/hip_bf16.h>


typedef __attribute__((ext_vector_type(8))) int v8i;
typedef __attribute__((ext_vector_type(4))) int v4i;
typedef __attribute__((ext_vector_type(2))) int v2i;

#define IN_F   1024
#define OUT_F  1024
#define TOKENS 1024
#define KSTEPS 16
#define PITCH  80

static __device__ __forceinline__ unsigned lds_off(const void* p) {
    return (unsigned)(size_t)p;
}

static __device__ __forceinline__ void async_copy16(unsigned lds_dst,
                                                    unsigned goff,
                                                    const signed char* base,
                                                    int imm_off) {
    asm volatile("global_load_async_to_lds_b128 %0, %1, %2 offset:%3"
                 :
                 : "v"(lds_dst), "v"(goff), "s"(base), "n"(imm_off)
                 : "memory");
}

static __device__ __forceinline__ void wait_async0() {
    asm volatile("s_wait_asynccnt 0" ::: "memory");
}

__global__ __launch_bounds__(256) void quantize_rows(
    const float* __restrict__ in, signed char* __restrict__ q,
    float* __restrict__ scale, int ncols)
{
    const int row = blockIdx.x;
    const float* rp = in + (size_t)row * ncols;
    __shared__ float red[256];

    float m = 0.f;
    for (int c = threadIdx.x; c < ncols; c += 256)
        m = fmaxf(m, fabsf(rp[c]));
    red[threadIdx.x] = m;
    __syncthreads();
    for (int s = 128; s > 0; s >>= 1) {
        if (threadIdx.x < s)
            red[threadIdx.x] = fmaxf(red[threadIdx.x], red[threadIdx.x + s]);
        __syncthreads();
    }
    const float sc = red[0] * (1.0f / 127.0f);
    if (threadIdx.x == 0) scale[row] = sc;
    const float inv = 1.0f / (sc + 1e-8f);
    for (int c = threadIdx.x; c < ncols; c += 256) {
        float v = rintf(rp[c] * inv);
        v = fminf(fmaxf(v, -128.0f), 127.0f);
        q[(size_t)row * ncols + c] = (signed char)v;
    }
}

__global__ __launch_bounds__(256, 2) void cim_int8_gemm(
    const signed char* __restrict__ xq, const signed char* __restrict__ wq,
    const float* __restrict__ sx, const float* __restrict__ sw,
    const float* __restrict__ bias, float* __restrict__ out)
{
    __shared__ signed char ldsA[2][128 * PITCH];
    __shared__ signed char ldsB[2][ 64 * PITCH];

    const int tid  = threadIdx.x;
    const int lane = tid & 31;
    const int wv   = tid >> 5;
    const int mw   = wv >> 1;
    const int nw   = wv & 1;
    const int lh   = lane >> 4;
    const int l16  = lane & 15;

    const int tBase = blockIdx.y * 128;
    const int oBase = blockIdx.x * 64;

    const int lrow = tid >> 2;
    const int lcol = (tid & 3) * 16;

    const unsigned oA0 = (unsigned)(tBase      + lrow) * IN_F + lcol;
    const unsigned oA1 = (unsigned)(tBase + 64 + lrow) * IN_F + lcol;
    const unsigned oB  = (unsigned)(oBase      + lrow) * IN_F + lcol;

    unsigned dA0[2], dA1[2], dB[2];
    #pragma unroll
    for (int b = 0; b < 2; ++b) {
        dA0[b] = lds_off(&ldsA[b][lrow        * PITCH + lcol]);
        dA1[b] = lds_off(&ldsA[b][(64 + lrow) * PITCH + lcol]);
        dB[b]  = lds_off(&ldsB[b][lrow        * PITCH + lcol]);
    }

    async_copy16(dA0[0], oA0, xq, 0);
    async_copy16(dA1[0], oA1, xq, 0);
    async_copy16(dB[0],  oB,  wq, 0);

    v8i acc[2][2] = {};

    wait_async0();
    __syncthreads();

    #pragma unroll
    for (int ks = 0; ks < KSTEPS; ++ks) {
        const int buf = ks & 1;

        if (ks < KSTEPS - 1) {
            const int k0 = (ks + 1) * 64;
            const int nb = buf ^ 1;
            async_copy16(dA0[nb], oA0, xq, k0);
            async_copy16(dA1[nb], oA1, xq, k0);
            async_copy16(dB[nb],  oB,  wq, k0);
        }

        v8i afrag[2];
        #pragma unroll
        for (int m16 = 0; m16 < 2; ++m16) {
            const signed char* ap =
                &ldsA[buf][(mw * 32 + m16 * 16 + l16) * PITCH + lh * 16];
            #pragma unroll
            for (int p = 0; p < 4; ++p) {
                v2i d = *(const v2i*)(ap + (p >> 1) * 32 + (p & 1) * 8);
                afrag[m16][2 * p]     = d.x;
                afrag[m16][2 * p + 1] = d.y;
            }
        }
        v8i bfrag[2];
        #pragma unroll
        for (int n16 = 0; n16 < 2; ++n16) {
            const signed char* bp =
                &ldsB[buf][(nw * 32 + n16 * 16 + l16) * PITCH + lh * 16];
            v4i lo = *(const v4i*)(bp);
            v4i hi = *(const v4i*)(bp + 32);
            bfrag[n16][0] = lo.x; bfrag[n16][1] = lo.y;
            bfrag[n16][2] = lo.z; bfrag[n16][3] = lo.w;
            bfrag[n16][4] = hi.x; bfrag[n16][5] = hi.y;
            bfrag[n16][6] = hi.z; bfrag[n16][7] = hi.w;
        }

        #pragma unroll
        for (int m16 = 0; m16 < 2; ++m16)
            #pragma unroll
            for (int n16 = 0; n16 < 2; ++n16)
                acc[m16][n16] = __builtin_amdgcn_wmma_i32_16x16x64_iu8(
                     true, afrag[m16],
                     true, bfrag[n16],
                    acc[m16][n16],  false,  false);

        if (ks < KSTEPS - 1) wait_async0();
        __syncthreads();
    }

    #pragma unroll
    for (int m16 = 0; m16 < 2; ++m16) {
        #pragma unroll
        for (int n16 = 0; n16 < 2; ++n16) {
            const int o   = oBase + nw * 32 + n16 * 16 + l16;
            const float swv = sw[o];
            const float bv  = bias[o];
            #pragma unroll
            for (int i = 0; i < 8; ++i) {
                const int t = tBase + mw * 32 + m16 * 16 + lh * 8 + i;
                out[(size_t)t * OUT_F + o] =
                    (float)acc[m16][n16][i] * sx[t] * swv + bv;
            }
        }
    }
}

extern "C" void kernel_launch(void* const* d_in, const int* in_sizes, int n_in,
                              void* d_out, int out_size, void* d_ws, size_t ws_size,
                              hipStream_t stream) {
    (void)in_sizes; (void)n_in; (void)out_size; (void)ws_size;
    const float* x    = (const float*)d_in[0];
    const float* w    = (const float*)d_in[1];
    const float* bias = (const float*)d_in[2];

    signed char* xq = (signed char*)d_ws;
    signed char* wq = xq + (size_t)TOKENS * IN_F;
    float*       sx = (float*)(wq + (size_t)OUT_F * IN_F);
    float*       sw = sx + TOKENS;
    float*       out = (float*)d_out;

    quantize_rows<<<TOKENS, 256, 0, stream>>>(x, xq, sx, IN_F);
    quantize_rows<<<OUT_F,  256, 0, stream>>>(w, wq, sw, IN_F);

    dim3 grid(OUT_F / 64, TOKENS / 128);
    cim_int8_gemm<<<grid, 256, 0, stream>>>(xq, wq, sx, sw, bias, out);
}
